// QMambaBlock_10960756539576
// MI455X (gfx1250) — compile-verified, matrix operand order checked by reading, not run on hardware
//
#include <hip/hip_runtime.h>

#define B_ 16
#define C_ 128
#define H_ 64
#define W_ 64
#define L_ (H_ * W_)   // 4096
#define S_ 16

typedef __attribute__((ext_vector_type(16))) __bf16 v16bf;
typedef __attribute__((ext_vector_type(8)))  __bf16 v8bf;
typedef __attribute__((ext_vector_type(8)))  float  v8f;

// K index inside a 32-wide chunk for element i of an A/B bf16 fragment.
// ISA 16-bit A 16x32 layout: lanes 0-15 -> K {0..7,16..23}, lanes 16-31 -> +8.
__device__ __forceinline__ int kpat(int i, int kg) {
    return (i & 7) + ((i >> 3) << 4) + (kg << 3);
}

__device__ __forceinline__ v16bf frag_from_row(const __bf16* rowp) {
    // rowp = &row[kg*8]; elements i<8 at rowp[i], i>=8 at rowp[16+i-8]
    v8bf lo = *(const v8bf*)(rowp);
    v8bf hi = *(const v8bf*)(rowp + 16);
    return __builtin_shufflevector(lo, hi, 0, 1, 2, 3, 4, 5, 6, 7,
                                   8, 9, 10, 11, 12, 13, 14, 15);
}

// ---------------------------------------------------------------------------
__global__ __launch_bounds__(256) void zero_f32_kernel(float* p, int n) {
    int i = blockIdx.x * 256 + threadIdx.x;
    if (i < n) p[i] = 0.f;
}

// Pre-pack conv weights to bf16 in exact WMMA B-fragment lane order:
// wpc[tap][kc][ntile][lane][i]
__global__ __launch_bounds__(256) void pack_convw_kernel(
    const float* __restrict__ conv_w, __bf16* __restrict__ wpc)
{
    int idx = blockIdx.x * 256 + threadIdx.x;
    if (idx >= 9 * 4 * 8 * 32 * 16) return;
    const int i    = idx & 15;
    const int lane = (idx >> 4) & 31;
    const int nt   = (idx >> 9) & 7;
    const int kc   = (idx >> 12) & 3;
    const int tap  = idx >> 14;
    const int n  = nt * 16 + (lane & 15);
    const int ci = kc * 32 + kpat(i, lane >> 4);
    wpc[idx] = (__bf16)conv_w[(n * C_ + ci) * 9 + tap];
}

// wpd[kc][ntile][lane][i]
__global__ __launch_bounds__(256) void pack_dtw_kernel(
    const float* __restrict__ dt_w, __bf16* __restrict__ wpd)
{
    int idx = blockIdx.x * 256 + threadIdx.x;
    if (idx >= 4 * 8 * 32 * 16) return;
    const int i    = idx & 15;
    const int lane = (idx >> 4) & 31;
    const int nt   = (idx >> 9) & 7;
    const int kc   = idx >> 12;
    const int n  = nt * 16 + (lane & 15);
    const int ci = kc * 32 + kpat(i, lane >> 4);
    wpd[idx] = (__bf16)dt_w[n * C_ + ci];
}

// ---------------------------------------------------------------------------
// 3x3 SAME conv as implicit GEMM on bf16 WMMA. Per block: 16 tokens x 128 co.
// Per K-chunk (32 ci) load one 3x18 halo patch to LDS, run all 9 taps from it.
// Fused bias + avg-pool partial sums.
// ---------------------------------------------------------------------------
__global__ __launch_bounds__(256) void conv3x3_wmma_kernel(
    const float* __restrict__ x, const __bf16* __restrict__ wpc,
    const float* __restrict__ conv_b, float* __restrict__ conv_out,
    float* __restrict__ pooled)
{
    __shared__ alignas(64) __bf16 patch[3][18][32];   // [row][w+1][ci]

    const int tid  = threadIdx.x;
    const int lane = tid & 31;
    const int wave = tid >> 5;

    const int m0 = blockIdx.x * 16;      // tokens share (b,h)
    const int b  = m0 >> 12;
    const int h  = (m0 >> 6) & 63;
    const int w0 = m0 & 63;

    const int nCol = wave * 16 + (lane & 15);
    const int kg   = lane >> 4;
    const int mrow = lane & 15;

    v8f acc = {};

    for (int kc = 0; kc < 4; ++kc) {
        const int cbase = kc * 32;
        __syncthreads();
        // halo patch fill: 3 rows x 18 cols x 32 ci = 1728 bf16
        for (int e = tid; e < 1728; e += 256) {
            const int kk = e & 31;
            const int t2 = e >> 5;       // 0..53
            const int j  = t2 % 18;
            const int r  = t2 / 18;
            const int hh = h + r - 1;
            const int ww = w0 + j - 1;
            float v = 0.f;
            if ((unsigned)hh < 64u && (unsigned)ww < 64u)
                v = x[((b * C_ + (cbase + kk)) * H_ + hh) * W_ + ww];
            patch[r][j][kk] = (__bf16)v;
        }
        __syncthreads();

#pragma unroll
        for (int tap = 0; tap < 9; ++tap) {
            const int dh = tap / 3;
            const int dw = tap % 3;
            const v16bf af = frag_from_row(&patch[dh][mrow + dw][kg * 8]);
            const v16bf bf = *(const v16bf*)(
                wpc + ((((tap * 4 + kc) * 8 + wave) * 32 + lane) << 4));
            acc = __builtin_amdgcn_wmma_f32_16x16x32_bf16(
                false, af, false, bf, (short)0, acc, false, false);
        }
    }

    const float bias = conv_b[nCol];
    float psum = 0.f;
#pragma unroll
    for (int r = 0; r < 8; ++r) {
        const int tloc = r + (kg << 3);      // D layout: M = r + 8*(lane>>4)
        const float v = acc[r] + bias;
        conv_out[((b * C_ + nCol) * H_ + h) * W_ + (w0 + tloc)] = v;
        psum += v;
    }
    atomicAdd(&pooled[b * C_ + nCol], psum);
}

// ---------------------------------------------------------------------------
// Spectral gate MLP: pooled/4096 -> g1 -> relu -> g2 -> sigmoid. 1 block per b.
__global__ __launch_bounds__(128) void gate_kernel(
    const float* __restrict__ pooled,
    const float* __restrict__ g1_w, const float* __restrict__ g1_b,
    const float* __restrict__ g2_w, const float* __restrict__ g2_b,
    float* __restrict__ g)
{
    __shared__ float sp[C_];
    __shared__ float sh[C_];
    const int b = blockIdx.x;
    const int o = threadIdx.x;
    sp[o] = pooled[b * C_ + o] * (1.f / (float)L_);
    __syncthreads();
    float a = g1_b[o];
    for (int c = 0; c < C_; ++c) a += sp[c] * g1_w[o * C_ + c];
    sh[o] = fmaxf(a, 0.f);
    __syncthreads();
    float a2 = g2_b[o];
    for (int c = 0; c < C_; ++c) a2 += sh[c] * g2_w[o * C_ + c];
    g[b * C_ + o] = 1.f / (1.f + __expf(-a2));
}

// In-place fp32 gate scale + write token-major bf16 copy x_seq[b][l][c]
__global__ __launch_bounds__(256) void gate_apply_kernel(
    float* __restrict__ xg, const float* __restrict__ g,
    __bf16* __restrict__ xg16)
{
    const int i = blockIdx.x * 256 + threadIdx.x;   // over B*C*L
    const int b = i >> 19;
    const int c = (i >> 12) & 127;
    const int l = i & 4095;
    const float v = xg[i] * g[(b << 7) | c];
    xg[i] = v;
    xg16[(((size_t)(b << 12) + l) << 7) | c] = (__bf16)v;
}

// ---------------------------------------------------------------------------
// dt = sigmoid(x_seq @ dt_w^T + dt_b), K=128 WMMA GEMM. A tile staged with
// CDNA5 async global->LDS copies (ASYNCcnt) from the bf16 token-major copy.
// ---------------------------------------------------------------------------
__global__ __launch_bounds__(256) void dt_wmma_kernel(
    const __bf16* __restrict__ xg16, const __bf16* __restrict__ wpd,
    const float* __restrict__ dt_b, float* __restrict__ dt_out)
{
    __shared__ alignas(64) __bf16 aT[16][32];   // [token][ci]

    const int tid  = threadIdx.x;
    const int lane = tid & 31;
    const int wave = tid >> 5;

    const int m0 = blockIdx.x * 16;
    const int b  = m0 >> 12;
    const int l0 = m0 & (L_ - 1);

    const int nCol = wave * 16 + (lane & 15);
    const int kg   = lane >> 4;
    const int mrow = lane & 15;

    v8f acc = {};
#pragma unroll
    for (int kc = 0; kc < 4; ++kc) {
        const int cbase = kc * 32;
        __syncthreads();
        {
            // 256 threads x 4B async copies = full 16x32 bf16 tile
            const int m  = tid >> 4;
            const int kp = tid & 15;
            const __bf16* gsrc =
                xg16 + ((size_t)(b * L_ + l0 + m) * C_ + cbase + kp * 2);
            unsigned ldsoff = (unsigned)(uintptr_t)(&aT[m][kp * 2]);
            unsigned long long ga = (unsigned long long)(uintptr_t)gsrc;
            asm volatile("global_load_async_to_lds_b32 %0, %1, off"
                         :: "v"(ldsoff), "v"(ga) : "memory");
        }
        asm volatile("s_wait_asynccnt 0x0" ::: "memory");
        __syncthreads();

        const v16bf af = frag_from_row(&aT[mrow][kg * 8]);
        const v16bf bf = *(const v16bf*)(
            wpd + (((kc * 8 + wave) * 32 + lane) << 4));
        acc = __builtin_amdgcn_wmma_f32_16x16x32_bf16(
            false, af, false, bf, (short)0, acc, false, false);
    }

    const float bias = dt_b[nCol];
#pragma unroll
    for (int r = 0; r < 8; ++r) {
        const int tloc = r + (kg << 3);
        const float v = acc[r] + bias;
        dt_out[(b * C_ + nCol) * L_ + l0 + tloc] = 1.f / (1.f + __expf(-v));
    }
}

// ---------------------------------------------------------------------------
// SSM scan: each 16-lane half-wave owns one (b,c); lane s owns state h_s.
// ---------------------------------------------------------------------------
__global__ __launch_bounds__(256) void ssm_scan_kernel(
    const float* __restrict__ xg, const float* __restrict__ dt,
    const float* __restrict__ A, const float* __restrict__ D,
    const float* __restrict__ identity, float* __restrict__ out)
{
    const int tid  = threadIdx.x;
    const int lane = tid & 31;
    const int wave = tid >> 5;
    const int grp  = lane >> 4;
    const int s    = lane & 15;
    const int pair = blockIdx.x * 16 + wave * 2 + grp;
    const int b    = pair >> 7;
    const int c    = pair & 127;

    const float An = -__expf(A[c * S_ + s]);
    const float Dp = __expf(D[c]);
    const long  base = (long)(b * C_ + c) * L_;

    float h = 0.f;
    for (int l = 0; l < L_; ++l) {
        if (((l & 511) == 0) && (l + 512 < L_)) {
            __builtin_prefetch(xg + base + l + 512, 0, 0);
            __builtin_prefetch(dt + base + l + 512, 0, 0);
            __builtin_prefetch(identity + base + l + 512, 0, 0);
        }
        const float xt  = xg[base + l];
        const float dtt = dt[base + l];
        h = h * __expf(An * dtt) + xt;
        float val = h * An;
        val += __shfl_xor(val, 1, 32);
        val += __shfl_xor(val, 2, 32);
        val += __shfl_xor(val, 4, 32);
        val += __shfl_xor(val, 8, 32);
        if (s == 0)
            out[base + l] = val + Dp * xt + identity[base + l];
    }
}

// ---------------------------------------------------------------------------
extern "C" void kernel_launch(void* const* d_in, const int* in_sizes, int n_in,
                              void* d_out, int out_size, void* d_ws, size_t ws_size,
                              hipStream_t stream)
{
    (void)in_sizes; (void)n_in; (void)out_size; (void)ws_size;

    const float* x      = (const float*)d_in[0];
    const float* conv_w = (const float*)d_in[1];
    const float* conv_b = (const float*)d_in[2];
    const float* dt_w   = (const float*)d_in[3];
    const float* dt_b   = (const float*)d_in[4];
    const float* A      = (const float*)d_in[5];
    const float* D      = (const float*)d_in[6];
    const float* g1_w   = (const float*)d_in[7];
    const float* g1_b   = (const float*)d_in[8];
    const float* g2_w   = (const float*)d_in[9];
    const float* g2_b   = (const float*)d_in[10];
    float* out = (float*)d_out;

    // workspace layout
    char* ws = (char*)d_ws;
    const size_t planeBytes = (size_t)B_ * C_ * L_ * sizeof(float);   // 32 MiB
    float*  conv_out = (float*)ws;                                    // [B,C,L]
    float*  dt_buf   = (float*)(ws + planeBytes);                     // [B,C,L]
    __bf16* xg16     = (__bf16*)(ws + 2 * planeBytes);                // [B,L,C] 16 MiB
    __bf16* wpc      = (__bf16*)(ws + 2 * planeBytes + 16777216);     // 288 KiB
    __bf16* wpd      = (__bf16*)(ws + 2 * planeBytes + 16777216 + 294912);
    float*  pooled   = (float*)(ws + 2 * planeBytes + 16777216 + 294912 + 32768);
    float*  g        = (float*)(ws + 2 * planeBytes + 16777216 + 294912 + 32768 + 8192);

    const int tiles = (B_ * H_ * W_) / 16;   // 4096 M-tiles

    zero_f32_kernel<<<(B_ * C_ + 255) / 256, 256, 0, stream>>>(pooled, B_ * C_);
    pack_convw_kernel<<<(9 * 4 * 8 * 32 * 16) / 256, 256, 0, stream>>>(conv_w, wpc);
    pack_dtw_kernel<<<(4 * 8 * 32 * 16) / 256, 256, 0, stream>>>(dt_w, wpd);
    conv3x3_wmma_kernel<<<tiles, 256, 0, stream>>>(x, wpc, conv_b, conv_out, pooled);
    gate_kernel<<<B_, 128, 0, stream>>>(pooled, g1_w, g1_b, g2_w, g2_b, g);
    gate_apply_kernel<<<(B_ * C_ * L_) / 256, 256, 0, stream>>>(conv_out, g, xg16);
    dt_wmma_kernel<<<tiles, 256, 0, stream>>>(xg16, wpd, dt_b, dt_buf);
    ssm_scan_kernel<<<(B_ * C_) / 16, 256, 0, stream>>>(conv_out, dt_buf, A, D, x, out);
}
